// HopfieldHead_23210003268292
// MI455X (gfx1250) — matrix operand order repaired and checked by reading, not compiled, not run on hardware
//
#include <hip/hip_runtime.h>

#define D_MODEL 1024
#define NPAT    8

// POD vector types (safe in unions, unlike HIP's ctor-wrapped float4 etc.)
typedef __attribute__((ext_vector_type(16))) __bf16       v16bf;
typedef __attribute__((ext_vector_type(8)))  float        v8f;
typedef __attribute__((ext_vector_type(4)))  float        f32x4;
typedef __attribute__((ext_vector_type(4)))  unsigned int u32x4;
typedef __attribute__((ext_vector_type(2)))  unsigned int u32x2;

union ABFrag { u32x4 u[2]; v16bf v; };   // 32 bytes = 16 bf16 per lane
union Pack4  { __bf16 b[4]; u32x2 u; };  // 4 bf16 packed for 8B stores

// Low 32 bits of a generic pointer to LDS == LDS byte offset (ISA 10.2).
__device__ __forceinline__ unsigned lds_addr(const void* p) {
    return (unsigned)(unsigned long long)p;
}

// gfx1250 async copy: HBM -> LDS with no VGPR round-trip, tracked by ASYNCcnt.
__device__ __forceinline__ void async_load_b128(unsigned lds, const void* gptr) {
    asm volatile("global_load_async_to_lds_b128 %0, %1, off"
                 :: "v"(lds), "v"(gptr) : "memory");
}
__device__ __forceinline__ void wait_async0() {
    asm volatile("s_wait_asynccnt 0x0" ::: "memory");
}

// ---------------------------------------------------------------------------
// Kernel 0: proj_w f32 -> bf16 (2 MB, negligible cost)
// ---------------------------------------------------------------------------
__global__ __launch_bounds__(256) void convert_w_bf16(
    const float* __restrict__ w, __bf16* __restrict__ wb)
{
    const int i = (blockIdx.x * 256 + threadIdx.x) * 4;  // D*D = 1<<20, exact
    f32x4 v = *(const f32x4*)(w + i);
    Pack4 p;
    p.b[0] = (__bf16)v.x; p.b[1] = (__bf16)v.y;
    p.b[2] = (__bf16)v.z; p.b[3] = (__bf16)v.w;
    *(u32x2*)(wb + i) = p.u;
}

// ---------------------------------------------------------------------------
// Kernel 1: fused scores -> softmax -> read; emit x = h + read as bf16.
// One wave32 per token; patterns staged in LDS (8*1024*4 = 32 KB).
// Streaming, memory-bound: reads h once, writes x once.
// ---------------------------------------------------------------------------
__global__ __launch_bounds__(256) void hopfield_stage1(
    const float* __restrict__ h,
    const float* __restrict__ patterns,
    __bf16* __restrict__ x,
    int n_tokens)
{
    __shared__ float sp[NPAT * D_MODEL];
    for (int i = threadIdx.x; i < NPAT * D_MODEL / 4; i += 256)
        ((f32x4*)sp)[i] = ((const f32x4*)patterns)[i];
    __syncthreads();

    const int wave  = threadIdx.x >> 5;
    const int lane  = threadIdx.x & 31;
    const int token = blockIdx.x * 8 + wave;
    if (token >= n_tokens) return;

    const f32x4* hrow = (const f32x4*)(h + (size_t)token * D_MODEL);

    f32x4 hv[8];                    // keep h in regs for the second pass
    float s[NPAT];
#pragma unroll
    for (int p = 0; p < NPAT; ++p) s[p] = 0.f;
#pragma unroll
    for (int j = 0; j < 8; ++j) {
        const int d4 = lane + j * 32;     // float4 index, 256 per row
        hv[j] = hrow[d4];
#pragma unroll
        for (int p = 0; p < NPAT; ++p) {
            f32x4 pv = ((const f32x4*)(sp + p * D_MODEL))[d4];
            s[p] += hv[j].x * pv.x + hv[j].y * pv.y +
                    hv[j].z * pv.z + hv[j].w * pv.w;
        }
    }
    // wave32 butterfly reduction: every lane ends with the full dot products
#pragma unroll
    for (int off = 16; off > 0; off >>= 1)
#pragma unroll
        for (int p = 0; p < NPAT; ++p)
            s[p] += __shfl_xor(s[p], off, 32);

    // softmax over 8 patterns (BETA = 1.0), replicated in all lanes
    float mx = s[0];
#pragma unroll
    for (int p = 1; p < NPAT; ++p) mx = fmaxf(mx, s[p]);
    float wsum = 0.f, wgt[NPAT];
#pragma unroll
    for (int p = 0; p < NPAT; ++p) { wgt[p] = __expf(s[p] - mx); wsum += wgt[p]; }
    const float inv = 1.0f / wsum;
#pragma unroll
    for (int p = 0; p < NPAT; ++p) wgt[p] *= inv;

    // x = h + sum_p wgt[p] * pattern[p], stored bf16 (halves GEMM input traffic)
    __bf16* xrow = x + (size_t)token * D_MODEL;
#pragma unroll
    for (int j = 0; j < 8; ++j) {
        const int d4 = lane + j * 32;
        f32x4 r = hv[j];
#pragma unroll
        for (int p = 0; p < NPAT; ++p) {
            f32x4 pv = ((const f32x4*)(sp + p * D_MODEL))[d4];
            r.x += wgt[p] * pv.x; r.y += wgt[p] * pv.y;
            r.z += wgt[p] * pv.z; r.w += wgt[p] * pv.w;
        }
        Pack4 pk;
        pk.b[0] = (__bf16)r.x; pk.b[1] = (__bf16)r.y;
        pk.b[2] = (__bf16)r.z; pk.b[3] = (__bf16)r.w;
        ((u32x2*)xrow)[d4] = pk.u;   // 8-byte store, element offset d4*4
    }
}

// ---------------------------------------------------------------------------
// Kernel 2: out[M,1024] = x_bf16 @ W_bf16^T + bias, f32 out, WMMA bf16.
// Block tile 256x128, BK=32, 8 waves in a 4x2 grid: each wave owns a
// 64x64 tile = 16 accum tiles (4 A-frags x 4 B-frags -> 16 WMMAs per
// 16 ds_load_b128, barriers amortized over 16 WMMAs). Double-buffered LDS
// fed by global_load_async_to_lds_b128 one K-tile ahead; s_wait_asynccnt 0
// + barrier publishes. LDS rows padded to 40 halves (80 B) -> 16 consecutive
// rows hit 16 distinct bank groups while keeping ds_load_b128 16-B aligned.
// ---------------------------------------------------------------------------
#define BM  256
#define BN  128
#define BK  32
#define LDP 40   // padded halves per LDS row

__global__ __launch_bounds__(256) void hopfield_gemm_wmma(
    const __bf16* __restrict__ X,   // [M, 1024] bf16
    const __bf16* __restrict__ W,   // [1024, 1024] bf16, row-major [e][d]
    const float*  __restrict__ bias,
    float* __restrict__ out,
    int M)
{
    __shared__ __bf16 lA[2][BM * LDP]; // 2 x 20 KB
    __shared__ __bf16 lB[2][BN * LDP]; // 2 x 10 KB

    const int tid  = threadIdx.x;
    const int wave = tid >> 5;
    const int lane = tid & 31;
    const int m0   = blockIdx.x * BM;
    const int n0   = blockIdx.y * BN;

    // async-loader mapping: chunk c -> row c>>2, halves [(c&3)*8, +8)
    const int rowL = tid >> 2;
    const int off  = (tid & 3) * 8;

    // wave tiling: 4x2 grid of waves; 64 rows x 64 cols per wave
    const int rowbase = (wave & 3) * 64;
    const int colbase = (wave >> 2) * 64;
    const int lr   = lane & 15;
    const int kh   = lane >> 4;          // 0: low K half, 1: high K half

    v8f acc[4][4];
#pragma unroll
    for (int i = 0; i < 4; ++i)
#pragma unroll
        for (int t = 0; t < 4; ++t) acc[i][t] = v8f{};

    // ---- issue one K-tile's async loads into buffer `buf` ----
    // A: 256 rows x 32 halves (1024 b128 chunks), B: 128 rows (512 chunks)
    auto issue_tile = [&](int buf, int k0) {
#pragma unroll
        for (int j = 0; j < 4; ++j) {
            const int row = rowL + j * 64;
            async_load_b128(lds_addr(&lA[buf][row * LDP + off]),
                            &X[(size_t)(m0 + row) * D_MODEL + k0 + off]);
        }
#pragma unroll
        for (int j = 0; j < 2; ++j) {
            const int row = rowL + j * 64;
            async_load_b128(lds_addr(&lB[buf][row * LDP + off]),
                            &W[(size_t)(n0 + row) * D_MODEL + k0 + off]);
        }
    };

    issue_tile(0, 0);
    wait_async0();
    __syncthreads();

    int cur = 0;
    for (int k0 = 0; k0 < D_MODEL; k0 += BK) {
        const int nxt = cur ^ 1;
        if (k0 + BK < D_MODEL)
            issue_tile(nxt, k0 + BK);    // overlaps with the WMMAs below

        // A fragments (16x32 bf16): lane<16 -> row lr, K {0..7}u{16..23};
        // lane>=16 -> row lr, K {8..15}u{24..31}
        ABFrag af[4];
        const int koff = kh * 8;
#pragma unroll
        for (int i = 0; i < 4; ++i) {
            const __bf16* ap = &lA[cur][(rowbase + i * 16 + lr) * LDP + koff];
            af[i].u[0] = *(const u32x4*)(ap);
            af[i].u[1] = *(const u32x4*)(ap + 16);
        }
        // 4 B fragments (32x16): lane<16 -> col lr, K {0..7}u{16..23};
        // lane>=16 -> col lr, K {8..15}u{24..31}. Each feeds 4 WMMAs.
        const int bk = kh * 8;
#pragma unroll
        for (int t = 0; t < 4; ++t) {
            ABFrag bf_;
            const __bf16* bp = &lB[cur][(colbase + t * 16 + lr) * LDP + bk];
            bf_.u[0] = *(const u32x4*)(bp);
            bf_.u[1] = *(const u32x4*)(bp + 16);
#pragma unroll
            for (int i = 0; i < 4; ++i)
                acc[i][t] = __builtin_amdgcn_wmma_f32_16x16x32_bf16(
                    false, af[i].v, false, bf_.v, (short)0, acc[i][t],
                    false, false);
        }

        wait_async0();      // my 6 async b128 for `nxt` have landed
        __syncthreads();    // everyone's landed + everyone done reading `cur`
        cur = nxt;
    }

    // Epilogue. C layout: VGPR r -> (M = r + 8*(lane>=16), N = lane&15)
#pragma unroll
    for (int i = 0; i < 4; ++i) {
        const int orow = m0 + rowbase + i * 16 + kh * 8;
#pragma unroll
        for (int t = 0; t < 4; ++t) {
            const int col = n0 + colbase + t * 16 + lr;
            const float b = bias[col];
#pragma unroll
            for (int r = 0; r < 8; ++r)
                out[(size_t)(orow + r) * D_MODEL + col] = acc[i][t][r] + b;
        }
    }
}

// ---------------------------------------------------------------------------
extern "C" void kernel_launch(void* const* d_in, const int* in_sizes, int n_in,
                              void* d_out, int out_size, void* d_ws, size_t ws_size,
                              hipStream_t stream)
{
    (void)n_in; (void)out_size; (void)ws_size;
    const float* h        = (const float*)d_in[0];
    const float* patterns = (const float*)d_in[1];
    const float* proj_w   = (const float*)d_in[2];
    const float* proj_b   = (const float*)d_in[3];
    float*       out      = (float*)d_out;

    const int n_tokens = in_sizes[0] / D_MODEL;          // 65536

    // workspace: x_bf16 [n_tokens,1024] then W_bf16 [1024,1024]
    __bf16* xb = (__bf16*)d_ws;
    __bf16* wb = xb + (size_t)n_tokens * D_MODEL;

    convert_w_bf16<<<dim3((D_MODEL * D_MODEL) / (256 * 4)), 256, 0, stream>>>(
        proj_w, wb);
    hopfield_stage1<<<dim3((n_tokens + 7) / 8), 256, 0, stream>>>(
        h, patterns, xb, n_tokens);
    hopfield_gemm_wmma<<<dim3(n_tokens / BM, D_MODEL / BN), 256, 0, stream>>>(
        xb, wb, proj_b, out, n_tokens);
}
